// LSTMStack_43344809951820
// MI455X (gfx1250) — matrix operand order repaired and checked by reading, not compiled, not run on hardware
//
#include <hip/hip_runtime.h>

typedef unsigned short u16;
typedef unsigned int   u32;
typedef __attribute__((ext_vector_type(16))) __bf16 v16bf;
typedef __attribute__((ext_vector_type(8)))  float  v8f;

union FragBF {
  v16bf v;
  uint4 q[2];
};

__device__ __forceinline__ u16 f2bf(float f) {
  u32 u = __float_as_uint(f);
  u32 r = u + 0x7FFFu + ((u >> 16) & 1u);   // round-to-nearest-even
  return (u16)(r >> 16);
}

__device__ __forceinline__ float sigm(float x) {
  return 1.0f / (1.0f + __expf(-x));
}

#define HSTATE 1024
#define GATES  4096
#define NB     64

__device__ __forceinline__ FragBF ld_frag32(const u16* __restrict__ p) {
  // 16 contiguous bf16 (32B) -> one fragment register pair
  FragBF f;
  f.q[0] = *(const uint4*)(p);
  f.q[1] = *(const uint4*)(p + 8);
  return f;
}

__device__ __forceinline__ FragBF ld_fragA(const u16* __restrict__ p) {
  // A-operand: two 8-element runs 16 elements apart (16-bit A layout)
  FragBF f;
  f.q[0] = *(const uint4*)(p);
  f.q[1] = *(const uint4*)(p + 16);
  return f;
}

// Fused dual-GEMM + LSTM cell for one timestep of one layer (all-bf16 inputs).
// gates(64x4096) = A1(64xK1) @ W1 + A2(64x1024) @ W2 + bias ; c,h = cell(gates).
// Weights are pre-transposed: WT[ncol][k], so B fragments are contiguous loads.
// Each workgroup: 16 h-columns across all 4 gate blocks; 8 waves;
// wave(gi,sub): gate gi, M-rows sub*32..sub*32+31 (two 16x16 M-tiles).
__global__ __launch_bounds__(256) void lstm_step_kernel(
    const u16*  __restrict__ A1, int K1, long a1_stride,
    const u16*  __restrict__ WT1,      // bf16 [4096][K1]
    const u16*  __restrict__ A2,       // bf16 h_prev [64][1024]
    const u16*  __restrict__ WT2,      // bf16 [4096][1024]
    const float* __restrict__ bias,    // 4096
    float* __restrict__ c_state,       // 64 x 1024 (running, in d_out tail)
    float* __restrict__ h_f32,         // 64 x 1024 (d_out tail)
    u16*   __restrict__ h_bf_out,      // 64 x 1024 bf16 (next-step operand)
    float* __restrict__ y_out,         // optional y[:, t, :] base (or null)
    long y_stride,
    int first)
{
  __shared__ float lds_g[NB * 4 * 16];      // gates [b][gate][hcol]

  const int tid    = threadIdx.x;
  const int wave   = tid >> 5;
  const int lane   = tid & 31;
  const int lane_n = lane & 15;
  const int half   = lane >> 4;
  const int gi     = wave & 3;              // gate block i/f/g/o
  const int sub    = wave >> 2;             // M-split
  const int hb     = blockIdx.x * 16;       // h-column base
  const int ncol   = gi * HSTATE + hb + lane_n;

  // per-lane streaming pointers (all 16B aligned)
  const u16* w1p = WT1 + (long)ncol * K1     + half * 8;
  const u16* w2p = WT2 + (long)ncol * HSTATE + half * 8;
  const int m0 = (sub * 2) * 16 + lane_n;
  const u16* a1p0 = A1 + (long)m0 * a1_stride + half * 8;
  const u16* a1p1 = a1p0 + 16 * a1_stride;
  const u16* a2p0 = A2 + (long)m0 * HSTATE + half * 8;
  const u16* a2p1 = a2p0 + 16 * HSTATE;

  v8f acc0 = {};
  v8f acc1 = {};

  // GEMM 1: A1 @ W1  (K1 is 512 or 1024, multiple of 32)
#pragma unroll 4
  for (int kk = 0; kk < K1; kk += 32) {
    FragBF b  = ld_fragA(w1p + kk);
    FragBF a0 = ld_fragA(a1p0 + kk);
    FragBF a1 = ld_fragA(a1p1 + kk);
    acc0 = __builtin_amdgcn_wmma_f32_16x16x32_bf16(false, a0.v, false, b.v,
                                                   (short)0, acc0, false, false);
    acc1 = __builtin_amdgcn_wmma_f32_16x16x32_bf16(false, a1.v, false, b.v,
                                                   (short)0, acc1, false, false);
  }
  // GEMM 2: A2 @ W2  (K = 1024)
#pragma unroll 4
  for (int kk = 0; kk < HSTATE; kk += 32) {
    FragBF b  = ld_fragA(w2p + kk);
    FragBF a0 = ld_fragA(a2p0 + kk);
    FragBF a1 = ld_fragA(a2p1 + kk);
    acc0 = __builtin_amdgcn_wmma_f32_16x16x32_bf16(false, a0.v, false, b.v,
                                                   (short)0, acc0, false, false);
    acc1 = __builtin_amdgcn_wmma_f32_16x16x32_bf16(false, a1.v, false, b.v,
                                                   (short)0, acc1, false, false);
  }

  // spill accumulator tiles (+bias) to LDS in [b][gate][hcol] order
  const float bv = bias[ncol];
#pragma unroll
  for (int j = 0; j < 8; ++j) {
    int mr0 = (sub * 2) * 16 + half * 8 + j;         // C layout: VGPR j -> M
    lds_g[(mr0 * 4 + gi) * 16 + lane_n] = acc0[j] + bv;
    int mr1 = (sub * 2 + 1) * 16 + half * 8 + j;
    lds_g[(mr1 * 4 + gi) * 16 + lane_n] = acc1[j] + bv;
  }
  __syncthreads();

  // fused LSTM cell: 64 batches x 16 h-cols, 4 elements per thread
#pragma unroll
  for (int i = 0; i < 4; ++i) {
    int e = tid + i * 256;
    int b_row = e >> 4;
    int j = e & 15;
    float iv = lds_g[(b_row * 4 + 0) * 16 + j];
    float fv = lds_g[(b_row * 4 + 1) * 16 + j];
    float gv = lds_g[(b_row * 4 + 2) * 16 + j];
    float ov = lds_g[(b_row * 4 + 3) * 16 + j];
    int hcol = hb + j;
    long idx = (long)b_row * HSTATE + hcol;
    float cp = first ? 0.0f : c_state[idx];
    float cn = sigm(fv) * cp + sigm(iv) * tanhf(gv);
    float hn = sigm(ov) * tanhf(cn);
    c_state[idx]  = cn;
    h_f32[idx]    = hn;
    h_bf_out[idx] = f2bf(hn);
    if (y_out) y_out[(long)b_row * y_stride + hcol] = hn;
  }
}

// LDS-tiled transpose + f32->bf16: dst[n][k] = cvt(src[k][n]), N = 4096.
__global__ __launch_bounds__(256) void transpose_cvt_kernel(
    const float* __restrict__ src, u16* __restrict__ dst, int K) {
  __shared__ u16 tile[32][33];
  const int kb = blockIdx.x * 32;
  const int nb = blockIdx.y * 32;
  const int r0 = threadIdx.x >> 5;
  const int c  = threadIdx.x & 31;
#pragma unroll
  for (int i = 0; i < 4; ++i) {
    int r = r0 + i * 8;
    tile[r][c] = f2bf(src[(long)(kb + r) * GATES + nb + c]);
  }
  __syncthreads();
#pragma unroll
  for (int i = 0; i < 4; ++i) {
    int r = r0 + i * 8;
    dst[(long)(nb + r) * K + kb + c] = tile[c][r];
  }
}

__global__ void cvt_f32_bf16_kernel(const float* __restrict__ s,
                                    u16* __restrict__ d, long n) {
  long i = (long)blockIdx.x * blockDim.x + threadIdx.x;
  long stride = (long)gridDim.x * blockDim.x;
  for (; i < n; i += stride) d[i] = f2bf(s[i]);
}

__global__ void zero_u16_kernel(u16* __restrict__ d, long n) {
  long i = (long)blockIdx.x * blockDim.x + threadIdx.x;
  long stride = (long)gridDim.x * blockDim.x;
  for (; i < n; i += stride) d[i] = 0;
}

extern "C" void kernel_launch(void* const* d_in, const int* in_sizes, int n_in,
                              void* d_out, int out_size, void* d_ws, size_t ws_size,
                              hipStream_t stream) {
  (void)in_sizes; (void)n_in; (void)out_size; (void)ws_size;
  const long B = 64, T = 1024, D = 512, H = 1024;

  const float* x   = (const float*)d_in[0];
  const float* Wx0 = (const float*)d_in[1];
  const float* Wh0 = (const float*)d_in[2];
  const float* b0  = (const float*)d_in[3];
  const float* Wx1 = (const float*)d_in[4];
  const float* Wh1 = (const float*)d_in[5];
  const float* b1  = (const float*)d_in[6];

  float* out = (float*)d_out;
  float* y  = out;                       // (B, T, H)
  float* c0 = out + B * T * H;
  float* h0 = c0 + B * H;
  float* c1 = h0 + B * H;
  float* h1 = c1 + B * H;

  // workspace: transposed bf16 weights + bf16 x + bf16 h ping-pong (~92.5 MB)
  u16* wx0t = (u16*)d_ws;                 // [4096][512]
  u16* wh0t = wx0t + (long)GATES * 512;   // [4096][1024]
  u16* wx1t = wh0t + (long)GATES * 1024;
  u16* wh1t = wx1t + (long)GATES * 1024;
  u16* xb   = wh1t + (long)GATES * 1024;  // [64][1024][512] bf16
  u16* h0b  = xb + B * T * D;             // 2 * [64][1024]
  u16* h1b  = h0b + 2 * B * H;

  // one-time (per call) preprocessing
  transpose_cvt_kernel<<<dim3(512 / 32, GATES / 32), 256, 0, stream>>>(Wx0, wx0t, 512);
  transpose_cvt_kernel<<<dim3(1024 / 32, GATES / 32), 256, 0, stream>>>(Wh0, wh0t, 1024);
  transpose_cvt_kernel<<<dim3(1024 / 32, GATES / 32), 256, 0, stream>>>(Wx1, wx1t, 1024);
  transpose_cvt_kernel<<<dim3(1024 / 32, GATES / 32), 256, 0, stream>>>(Wh1, wh1t, 1024);
  cvt_f32_bf16_kernel<<<8192, 256, 0, stream>>>(x, xb, B * T * D);
  zero_u16_kernel<<<256, 256, 0, stream>>>(h0b, 4 * B * H);

  for (int t = 0; t < (int)T; ++t) {
    int p = t & 1;
    int first = (t == 0) ? 1 : 0;
    // layer 0: gates = x[:,t,:] @ Wx0 + h0_prev @ Wh0 + b0
    lstm_step_kernel<<<64, 256, 0, stream>>>(
        xb + (long)t * D, /*K1=*/(int)D, /*a1_stride=*/T * D,
        wx0t, h0b + (long)p * B * H, wh0t, b0,
        c0, h0, h0b + (long)(p ^ 1) * B * H,
        /*y_out=*/nullptr, 0, first);
    // layer 1: gates = h0_new @ Wx1 + h1_prev @ Wh1 + b1 ; y[:,t,:] = h1_new
    lstm_step_kernel<<<64, 256, 0, stream>>>(
        h0b + (long)(p ^ 1) * B * H, /*K1=*/(int)H, /*a1_stride=*/H,
        wx1t, h1b + (long)p * B * H, wh1t, b1,
        c1, h1, h1b + (long)(p ^ 1) * B * H,
        /*y_out=*/y + (long)t * H, /*y_stride=*/T * H, first);
  }
}
